// myZCANorm_76836964925863
// MI455X (gfx1250) — matrix operand order repaired and checked by reading, not compiled, not run on hardware
//
#include <hip/hip_runtime.h>
#include <hip/hip_bf16.h>

// ---- constants for this problem ----
#define NN   64
#define CC   256
#define MM   65536          // N*H*W
#define EPSF 1e-5f
#define NEIG 32
#define NPOW 10

typedef __attribute__((ext_vector_type(16))) __bf16 v16bf;
typedef __attribute__((ext_vector_type(8)))  __bf16 v8bf;
typedef __attribute__((ext_vector_type(8)))  float  v8f;

__device__ __forceinline__ unsigned short f2bf(float f) {
    union { float f; unsigned u; } u; u.f = f;
    unsigned r = u.u + 0x7FFFu + ((u.u >> 16) & 1u);   // round-to-nearest-even
    return (unsigned short)(r >> 16);
}

__device__ __forceinline__ v16bf mk_a(const __bf16* p, int kb) {
    v8bf lo = *(const v8bf*)(p + kb);
    v8bf hi = *(const v8bf*)(p + kb + 8);
    v16bf a;
#pragma unroll
    for (int t = 0; t < 8; ++t) { a[t] = lo[t]; a[t + 8] = hi[t]; }
    return a;
}

// ---------------- kernel 1: per-channel mean / rstd ----------------
__global__ __launch_bounds__(256)
void zca_stats(const float* __restrict__ x, float* __restrict__ mu,
               float* __restrict__ rstd) {
    __shared__ float reds[256], redq[256];
    const int c = blockIdx.x;
    const int t = threadIdx.x;
    float s = 0.f, q = 0.f;
    for (int i = 0; i < MM / 256; ++i) {
        int m = t + i * 256;
        int n = m >> 10;
        float v = x[((size_t)n * CC + c) * 1024 + (m & 1023)];
        s += v; q += v * v;
    }
    reds[t] = s; redq[t] = q; __syncthreads();
    for (int k = 128; k > 0; k >>= 1) {
        if (t < k) { reds[t] += reds[t + k]; redq[t] += redq[t + k]; }
        __syncthreads();
    }
    if (t == 0) {
        float sum = reds[0], sq = redq[0];
        float var = (sq - sum * sum / (float)MM) / (float)(MM - 1); // ddof=1
        mu[c]   = sum / (float)MM;
        rstd[c] = rsqrtf(var + EPSF);
    }
}

// ---------------- kernel 2: xs = (x - mu)*rstd -> bf16 [C, M] ----------------
__global__ __launch_bounds__(256)
void zca_normalize(const float* __restrict__ x, const float* __restrict__ mu,
                   const float* __restrict__ rstd, unsigned short* __restrict__ xs) {
    const size_t total = (size_t)NN * CC * 1024;     // 16M
    size_t gid = (size_t)blockIdx.x * blockDim.x + threadIdx.x;
    const size_t stride = (size_t)gridDim.x * blockDim.x;
    for (size_t i = gid; i < total; i += stride) {
        int n = (int)(i >> 18);
        int c = (int)((i >> 10) & 255);
        int w = (int)(i & 1023);
        int m = (n << 10) | w;
        xs[(size_t)c * MM + m] = f2bf((x[i] - mu[c]) * rstd[c]);
    }
}

// ---------------- kernel 2b: xsT[m][c] = xs[c][m]  (LDS tile transpose) ----------------
__global__ __launch_bounds__(256)
void zca_transpose(const unsigned short* __restrict__ xs,
                   unsigned short* __restrict__ xsT) {
    __shared__ __attribute__((aligned(64))) unsigned short tile[16 * 272]; // pad stride
    const int c0 = blockIdx.y * 16;
    const int m0 = blockIdx.x * 256;
    const int t  = threadIdx.x;
    {   // coalesced load: 16 rows x 512B
        int c = t >> 4, moff = (t & 15) * 16;
        v16bf val = *(const v16bf*)(xs + (size_t)(c0 + c) * MM + m0 + moff);
        *(v16bf*)&tile[c * 272 + moff] = val;
    }
    __syncthreads();
    {   // each thread emits one m-row of 16 channels (32B contiguous)
        int m = t;
        v16bf v;
#pragma unroll
        for (int h = 0; h < 16; ++h) v[h] = ((const __bf16*)tile)[h * 272 + m];
        *(v16bf*)(xsT + (size_t)(m0 + m) * CC + c0) = v;
    }
}

// ---------------- kernel 3a: xxt = eps * I ----------------
__global__ __launch_bounds__(256)
void zca_init_xxt(float* __restrict__ xxt) {
    int i = blockIdx.x * 256 + threadIdx.x;
    xxt[i] = ((i >> 8) == (i & 255)) ? EPSF : 0.f;
}

// ---- kernel 3b: covariance, WMMA bf16, 2x2 macro-tiles + symmetry + split-K ----
__global__ __launch_bounds__(256)
void zca_cov(const __bf16* __restrict__ xs, float* __restrict__ xxt) {
    const int wid  = (int)((blockIdx.x * blockDim.x + threadIdx.x) >> 5); // 0..2303
    const int lane = threadIdx.x & 31;
    const int chunk = wid & 63;           // 64 K-chunks of 1024
    int p = wid >> 6;                     // 0..35 upper-triangular macro-tiles (8x8)
    int tI = 0;
    while (p >= (8 - tI)) { p -= (8 - tI); ++tI; }
    const int tJ = tI + p;
    const int r0 = tI * 32, c0 = tJ * 32;
    const int k0 = chunk * 1024;

    const int row = lane & 15;
    const int kb  = (lane >> 4) * 16;
    const int kbB = (lane >> 4) * 16;
    const __bf16* pA0 = xs + (size_t)(r0 + row) * MM;
    const __bf16* pA1 = xs + (size_t)(r0 + 16 + row) * MM;
    const __bf16* pB0 = xs + (size_t)(c0 + row) * MM;
    const __bf16* pB1 = xs + (size_t)(c0 + 16 + row) * MM;

    v8f a00 = {}, a01 = {}, a10 = {}, a11 = {};
    for (int k = k0; k < k0 + 1024; k += 32) {
        v16bf A0 = mk_a(pA0 + k, kb);
        v16bf A1 = mk_a(pA1 + k, kb);
        v16bf B0 = *(const v16bf*)(pB0 + k + kbB);
        v16bf B1 = *(const v16bf*)(pB1 + k + kbB);
        a00 = __builtin_amdgcn_wmma_f32_16x16x32_bf16(false, A0, false, B0, (short)0, a00, false, false);
        a01 = __builtin_amdgcn_wmma_f32_16x16x32_bf16(false, A0, false, B1, (short)0, a01, false, false);
        a10 = __builtin_amdgcn_wmma_f32_16x16x32_bf16(false, A1, false, B0, (short)0, a10, false, false);
        a11 = __builtin_amdgcn_wmma_f32_16x16x32_bf16(false, A1, false, B1, (short)0, a11, false, false);
    }

    const float invM1 = 1.0f / (float)(MM - 1);
    const int colD  = lane & 15;
    const int rbase = (lane >> 4) * 8;
    const bool mirror = (tI != tJ);
    auto emit = [&](v8f acc, int rA, int cB) {
#pragma unroll
        for (int v = 0; v < 8; ++v) {
            float val = acc[v] * invM1;
            int r = rA + rbase + v, cc = cB + colD;
            atomicAdd(&xxt[r * 256 + cc], val);
            if (mirror) atomicAdd(&xxt[cc * 256 + r], val);
        }
    };
    emit(a00, r0,      c0);
    emit(a01, r0,      c0 + 16);
    emit(a10, r0 + 16, c0);
    emit(a11, r0 + 16, c0 + 16);
}

// ---------------- kernel 4: power iteration + deflation (one block) ----------------
__global__ __launch_bounds__(256)
void zca_power(float* __restrict__ A, float* __restrict__ V,
               float* __restrict__ scale) {
    __shared__ float vsh[256], red[256];
    const int c = threadIdx.x;
    float stopped = 0.f;

    for (int e = 0; e < NEIG; ++e) {
        vsh[c] = 1.0f; __syncthreads();
        for (int it = 0; it < NPOW; ++it) {
            float s = 0.f;
            for (int k = 0; k < 256; ++k) s = fmaf(A[c * 256 + k], vsh[k], s);
            red[c] = s * s; __syncthreads();
            for (int k = 128; k > 0; k >>= 1) {
                if (c < k) red[c] += red[c + k];
                __syncthreads();
            }
            float vn = s / fmaxf(sqrtf(red[0]), EPSF);
            __syncthreads();
            vsh[c] = vn; __syncthreads();
        }
        float w = 0.f;
        for (int k = 0; k < 256; ++k) w = fmaf(A[c * 256 + k], vsh[k], w);
        red[c] = w / vsh[c]; __syncthreads();
        for (int k = 128; k > 0; k >>= 1) {
            if (c < k) red[c] += red[c + k];
            __syncthreads();
        }
        float lam = red[0] / 256.0f;
        __syncthreads();
        int valid = (lam >= 0.f) && (stopped == 0.f);
        if (valid) {
            for (int k = 0; k < 256; ++k) A[c * 256 + k] -= w * vsh[k];
        }
        if (lam < 0.f) stopped = 1.f;
        V[e * 256 + c] = vsh[c];
        if (c == 0) scale[e] = valid ? rsqrtf(lam + EPSF) : 0.0f;
        __syncthreads();
    }
}

// ---------------- kernel 5: P = V^T diag(scale) V -> bf16 [256,256] ----------------
__global__ __launch_bounds__(256)
void zca_pmat(const float* __restrict__ V, const float* __restrict__ scale,
              unsigned short* __restrict__ Pb) {
    const int i = blockIdx.x, j = threadIdx.x;
    float p = 0.f;
#pragma unroll
    for (int k = 0; k < NEIG; ++k)
        p = fmaf(scale[k] * V[k * 256 + i], V[k * 256 + j], p);
    Pb[i * 256 + j] = f2bf(p);
}

// ---- kernel 6: out = weight * (P @ xs) + bias; one wave = one m-tile, all 256 ch ----
__global__ __launch_bounds__(256)
void zca_project(const __bf16* __restrict__ Pb, const __bf16* __restrict__ xsT,
                 const float* __restrict__ weight, const float* __restrict__ bias,
                 float* __restrict__ out) {
    const int wid  = (int)((blockIdx.x * blockDim.x + threadIdx.x) >> 5); // 0..4095
    const int lane = threadIdx.x & 31;
    const int m0  = wid * 16;
    const int row = lane & 15;
    const int kb  = (lane >> 4) * 16;
    const int kbB = (lane >> 4) * 16;

    v8f acc[16];
#pragma unroll
    for (int tc = 0; tc < 16; ++tc) acc[tc] = (v8f){};

    // B columns: contiguous 32B per lane from xsT (m-row holds all 256 channels)
    const __bf16* pB = xsT + (size_t)(m0 + (lane & 15)) * 256 + kbB;

    for (int j = 0; j < 8; ++j) {               // K = 256 in steps of 32
        v16bf b = *(const v16bf*)(pB + j * 32);
#pragma unroll
        for (int tc = 0; tc < 16; ++tc) {       // all channel tiles share this B
            const __bf16* pA = Pb + (size_t)(tc * 16 + row) * 256 + j * 32;
            v16bf a = mk_a(pA, kb);
            acc[tc] = __builtin_amdgcn_wmma_f32_16x16x32_bf16(
                false, a, false, b, (short)0, acc[tc], false, false);
        }
    }

    const int m = m0 + (lane & 15);
    const int n = m >> 10, mw = m & 1023;
    const int rbase = (lane >> 4) * 8;
    float* obase = out + (size_t)n * CC * 1024 + mw;
#pragma unroll
    for (int tc = 0; tc < 16; ++tc) {
#pragma unroll
        for (int v = 0; v < 8; ++v) {
            int ch = tc * 16 + rbase + v;
            obase[(size_t)ch * 1024] = acc[tc][v] * weight[ch] + bias[ch];
        }
    }
}

// ---------------- host launcher ----------------
extern "C" void kernel_launch(void* const* d_in, const int* in_sizes, int n_in,
                              void* d_out, int out_size, void* d_ws, size_t ws_size,
                              hipStream_t stream) {
    const float* x      = (const float*)d_in[0];
    const float* weight = (const float*)d_in[1];
    const float* bias   = (const float*)d_in[2];
    float*       out    = (float*)d_out;

    char* ws = (char*)d_ws;
    size_t off = 0;
    unsigned short* xs  = (unsigned short*)(ws + off); off += (size_t)CC * MM * 2; // 32 MB
    unsigned short* xsT = (unsigned short*)(ws + off); off += (size_t)CC * MM * 2; // 32 MB
    float* xxt   = (float*)(ws + off); off += (size_t)CC * CC * 4;                 // 256 KB
    float* V     = (float*)(ws + off); off += (size_t)NEIG * CC * 4;               // 32 KB
    float* scale = (float*)(ws + off); off += 64 * 4;
    float* mu    = (float*)(ws + off); off += CC * 4;
    float* rstd  = (float*)(ws + off); off += CC * 4;
    unsigned short* Pb = (unsigned short*)(ws + off); off += (size_t)CC * CC * 2;  // 128 KB

    zca_stats    <<<CC,    256, 0, stream>>>(x, mu, rstd);
    zca_normalize<<<16384, 256, 0, stream>>>(x, mu, rstd, xs);
    zca_transpose<<<dim3(MM / 256, CC / 16), 256, 0, stream>>>(xs, xsT);
    zca_init_xxt <<<CC,    256, 0, stream>>>(xxt);
    zca_cov      <<<288,   256, 0, stream>>>((const __bf16*)xs, xxt);   // 2304 waves
    zca_power    <<<1,     256, 0, stream>>>(xxt, V, scale);
    zca_pmat     <<<CC,    256, 0, stream>>>(V, scale, Pb);
    zca_project  <<<512,   256, 0, stream>>>((const __bf16*)Pb, (const __bf16*)xsT,
                                             weight, bias, out);        // 4096 waves
}
